// Validator_32813550142007
// MI455X (gfx1250) — matrix operand order repaired and checked by reading, not compiled, not run on hardware
//
#include <hip/hip_runtime.h>
#include <hip/hip_bf16.h>
#include <math.h>

typedef __attribute__((ext_vector_type(16))) __bf16 v16bf;
typedef __attribute__((ext_vector_type(8)))  __bf16 v8bf;
typedef __attribute__((ext_vector_type(4)))  __bf16 v4bf;
typedef __attribute__((ext_vector_type(8)))  float  v8f;

namespace {
constexpr int   kV    = 50257;
constexpr int   kD    = 1024;
constexpr int   kH    = 8;
constexpr int   kHid  = 4096;
constexpr int   kB    = 4;
constexpr int   kS    = 512;
constexpr int   kTopK = 8;
constexpr int   kNact = 512;
constexpr int   kM    = kB * kS;          // 2048 token rows
constexpr float kSqrtD = 32.0f;
constexpr float kImp   = 3.0f;

// GEMM tiling: block = 256 threads = 8 wave32; block tile 128(M) x 128(N) x 32(K)
// waves arranged 4(M) x 2(N); each wave owns 32x64 -> 2 A frags, 4 B frags,
// 8 WMMAs per K-step with each B fragment feeding two WMMAs.
constexpr int TM = 128, TN = 128, TK = 32;
constexpr int ASTR = 40;    // A LDS row stride (halves): 80B/row, 16B aligned
constexpr int BSTR = 136;   // B LDS row stride (halves): 272B/row, 16B aligned
}

// ---------------------------------------------------------------------------
// Generic strided-batched GEMM:  C = act(alpha * A @ op(B) + bias + residual)
//   A: M x K (row-major, lda);  B: flags&4 ? K x N : N x K (i.e. default A@W^T)
//   z decomposed as (zb = z/Hdiv, zh = z%Hdiv) with independent strides.
//   flags: 1 = ReLU, 2 = add residual R (C-shaped), 4 = B is K-major
// ---------------------------------------------------------------------------
__global__ __launch_bounds__(256) void gemm_wmma_kernel(
    const float* __restrict__ A, const float* __restrict__ Bm,
    const float* __restrict__ bias, const float* __restrict__ R,
    float* __restrict__ C,
    int M, int N, int K, int lda, int ldb, int ldc,
    long sAb, long sAh, long sBb, long sBh, long sCb, long sCh,
    int Hdiv, float alpha, int flags)
{
  __shared__ __bf16 As[2][TM][ASTR];
  __shared__ __bf16 Bs[2][TK][BSTR];

  const int tid  = threadIdx.x;
  const int lane = tid & 31;
  const int wave = tid >> 5;
  const int zb = (int)blockIdx.z / Hdiv;
  const int zh = (int)blockIdx.z % Hdiv;

  const float* Ab = A  + (long)zb * sAb + (long)zh * sAh;
  const float* Bb = Bm + (long)zb * sBb + (long)zh * sBh;
  float*       Cb = C  + (long)zb * sCb + (long)zh * sCh;
  const float* Rb = (flags & 2) ? (R + (long)zb * sCb + (long)zh * sCh) : nullptr;

  const int m0 = blockIdx.y * TM;
  const int n0 = blockIdx.x * TN;
  const bool tb = (flags & 4) != 0;

  const int hi  = lane >> 4;                // per ISA A/C VGPR layouts
  const int l15 = lane & 15;
  const int wm  = (wave >> 1) * 32;         // wave M origin within tile
  const int wn  = (wave & 1) * 64;          // wave N origin within tile

  v8f acc[8] = {};          // acc[nt] = rows wm..wm+15, acc[4+nt] = rows wm+16..wm+31
  float4 ra[4], rb[4];      // staging registers (pipeline)

  // ---- issue next tile's global loads (coalesced float4 = global_load_b128)
  auto load_regs = [&](int k0) {
    #pragma unroll
    for (int e = 0; e < 4; ++e) {
      int fidx = e * 256 + tid;             // 0..1023
      int r = fidx >> 3, cg = fidx & 7;
      ra[e] = *(const float4*)(Ab + (long)(m0 + r) * lda + k0 + cg * 4);
    }
    if (tb) {
      #pragma unroll
      for (int e = 0; e < 4; ++e) {
        int fidx = e * 256 + tid;
        int kk = fidx >> 5, ng = fidx & 31;
        rb[e] = *(const float4*)(Bb + (long)(k0 + kk) * ldb + n0 + ng * 4);
      }
    } else {
      #pragma unroll
      for (int e = 0; e < 4; ++e) {
        int fidx = e * 256 + tid;
        int n = fidx >> 3, kg = fidx & 7;
        rb[e] = *(const float4*)(Bb + (long)(n0 + n) * ldb + k0 + kg * 4);
      }
    }
  };

  // ---- convert f32->bf16 and commit staged registers into LDS buffer p
  auto store_tile = [&](int p) {
    #pragma unroll
    for (int e = 0; e < 4; ++e) {
      int fidx = e * 256 + tid;
      int r = fidx >> 3, cg = fidx & 7;
      v4bf h; h[0] = (__bf16)ra[e].x; h[1] = (__bf16)ra[e].y;
              h[2] = (__bf16)ra[e].z; h[3] = (__bf16)ra[e].w;
      *(v4bf*)&As[p][r][cg * 4] = h;
    }
    if (tb) {
      #pragma unroll
      for (int e = 0; e < 4; ++e) {
        int fidx = e * 256 + tid;
        int kk = fidx >> 5, ng = fidx & 31;
        v4bf h; h[0] = (__bf16)rb[e].x; h[1] = (__bf16)rb[e].y;
                h[2] = (__bf16)rb[e].z; h[3] = (__bf16)rb[e].w;
        *(v4bf*)&Bs[p][kk][ng * 4] = h;
      }
    } else {
      #pragma unroll
      for (int e = 0; e < 4; ++e) {
        int fidx = e * 256 + tid;
        int n = fidx >> 3, kg = fidx & 7;
        Bs[p][kg * 4 + 0][n] = (__bf16)rb[e].x;
        Bs[p][kg * 4 + 1][n] = (__bf16)rb[e].y;
        Bs[p][kg * 4 + 2][n] = (__bf16)rb[e].z;
        Bs[p][kg * 4 + 3][n] = (__bf16)rb[e].w;
      }
    }
  };

  // ---- 8 WMMAs from LDS buffer p; each B fragment feeds two WMMAs
  auto compute = [&](int p) {
    // A fragments: lane<16 -> K{0..7,16..23}, lane>=16 -> K{8..15,24..31}
    v8bf alo0 = *(const v8bf*)&As[p][wm + l15][hi * 8];
    v8bf ahi0 = *(const v8bf*)&As[p][wm + l15][16 + hi * 8];
    v8bf alo1 = *(const v8bf*)&As[p][wm + 16 + l15][hi * 8];
    v8bf ahi1 = *(const v8bf*)&As[p][wm + 16 + l15][16 + hi * 8];
    v16bf a0, a1;
    #pragma unroll
    for (int i = 0; i < 8; ++i) {
      a0[i] = alo0[i]; a0[i + 8] = ahi0[i];
      a1[i] = alo1[i]; a1[i + 8] = ahi1[i];
    }
    #pragma unroll
    for (int nt = 0; nt < 4; ++nt) {
      v16bf bf;
      #pragma unroll
      for (int i = 0; i < 8; ++i) {
        bf[i]     = Bs[p][hi * 8 + i][wn + nt * 16 + l15];
        bf[i + 8] = Bs[p][16 + hi * 8 + i][wn + nt * 16 + l15];
      }
      acc[nt]     = __builtin_amdgcn_wmma_f32_16x16x32_bf16(
          false, a0, false, bf, (short)0, acc[nt],     false, false);
      acc[4 + nt] = __builtin_amdgcn_wmma_f32_16x16x32_bf16(
          false, a1, false, bf, (short)0, acc[4 + nt], false, false);
    }
  };

  const bool fast = (m0 + TM <= M) && (n0 + TN <= N) && ((K & (TK - 1)) == 0);
  if (fast) {
    // -------- software-pipelined double-buffered main loop --------
    load_regs(0);
    store_tile(0);
    __syncthreads();
    int p = 0;
    for (int k0 = 0; k0 < K; k0 += TK) {
      const bool more = (k0 + TK < K);
      if (more) {
        load_regs(k0 + TK);                 // global loads fly under the WMMAs
        if (k0 + 2 * TK < K) {              // L2 prefetch of the k+2 slab
          __builtin_prefetch(Ab + (long)(m0 + (tid >> 1)) * lda + k0 + 2 * TK + (tid & 1) * 16, 0, 1);
          if (!tb)
            __builtin_prefetch(Bb + (long)(n0 + (tid >> 1)) * ldb + k0 + 2 * TK + (tid & 1) * 16, 0, 1);
        }
      }
      compute(p);
      if (more) store_tile(p ^ 1);
      __syncthreads();
      p ^= 1;
    }
  } else {
    // -------- guarded edge loop (only the ragged dec_W N-block) --------
    for (int k0 = 0; k0 < K; k0 += TK) {
      #pragma unroll
      for (int e = 0; e < 16; ++e) {
        int idx = e * 256 + tid;            // 0..4095
        int r = idx >> 5, c = idx & 31;
        int gr = m0 + r, gc = k0 + c;
        float v = (gr < M && gc < K) ? Ab[(long)gr * lda + gc] : 0.0f;
        As[0][r][c] = (__bf16)v;
      }
      #pragma unroll
      for (int e = 0; e < 16; ++e) {
        int idx = e * 256 + tid;
        int kk, nn;
        if (tb) { nn = idx & 127; kk = idx >> 7; }
        else    { kk = idx & 31;  nn = idx >> 5; }
        int gn = n0 + nn, gk = k0 + kk;
        float v;
        if (tb) v = (gk < K && gn < N) ? Bb[(long)gk * ldb + gn] : 0.0f;
        else    v = (gn < N && gk < K) ? Bb[(long)gn * ldb + gk] : 0.0f;
        Bs[0][kk][nn] = (__bf16)v;
      }
      __syncthreads();
      compute(0);
      __syncthreads();
    }
  }

  // ---- epilogue: D layout VGPR r -> M = r + 8*hi, N = lane&15 ----
  #pragma unroll
  for (int nt = 0; nt < 4; ++nt) {
    int col = n0 + wn + nt * 16 + l15;
    if (col >= N) continue;
    #pragma unroll
    for (int half = 0; half < 2; ++half) {
      #pragma unroll
      for (int r = 0; r < 8; ++r) {
        int row = m0 + wm + half * 16 + hi * 8 + r;
        if (row >= M) continue;
        float v = alpha * acc[half * 4 + nt][r];
        if (bias) v += bias[col];
        if (Rb)   v += Rb[(long)row * ldc + col];
        if (flags & 1) v = fmaxf(v, 0.0f);
        Cb[(long)row * ldc + col] = v;
      }
    }
  }
}

// ---------------------------------------------------------------------------
// Elementwise / reduction kernels
// ---------------------------------------------------------------------------
__global__ void embed_kernel(const int* __restrict__ inputs,
                             const float* __restrict__ emb,
                             float* __restrict__ x) {
  long i = (long)blockIdx.x * 256 + threadIdx.x;
  if (i >= (long)kM * kD) return;
  long t = i / kD; int d = (int)(i % kD);
  x[i] = emb[(long)inputs[t] * kD + d];
}

__global__ __launch_bounds__(256) void layernorm_kernel(
    const float* __restrict__ X, const float* __restrict__ g,
    const float* __restrict__ b, float* __restrict__ Y) {
  const long row = blockIdx.x;
  const float* x = X + row * kD;
  float* y = Y + row * kD;
  __shared__ float red[256];
  const int t = threadIdx.x;
  float s = 0.0f;
  for (int i = t; i < kD; i += 256) s += x[i];
  red[t] = s; __syncthreads();
  for (int o = 128; o > 0; o >>= 1) { if (t < o) red[t] += red[t + o]; __syncthreads(); }
  const float mean = red[0] / kD; __syncthreads();
  float v = 0.0f;
  for (int i = t; i < kD; i += 256) { float d = x[i] - mean; v += d * d; }
  red[t] = v; __syncthreads();
  for (int o = 128; o > 0; o >>= 1) { if (t < o) red[t] += red[t + o]; __syncthreads(); }
  const float inv = rsqrtf(red[0] / kD + 1e-5f);
  for (int i = t; i < kD; i += 256) y[i] = (x[i] - mean) * inv * g[i] + b[i];
}

__global__ __launch_bounds__(256) void softmax_kernel(float* __restrict__ X, int L) {
  float* x = X + (long)blockIdx.x * L;
  __shared__ float red[256];
  const int t = threadIdx.x;
  float m = -3.0e38f;
  for (int i = t; i < L; i += 256) m = fmaxf(m, x[i]);
  red[t] = m; __syncthreads();
  for (int o = 128; o > 0; o >>= 1) { if (t < o) red[t] = fmaxf(red[t], red[t + o]); __syncthreads(); }
  const float mx = red[0]; __syncthreads();
  float s = 0.0f;
  for (int i = t; i < L; i += 256) { float e = __expf(x[i] - mx); x[i] = e; s += e; }
  red[t] = s; __syncthreads();
  for (int o = 128; o > 0; o >>= 1) { if (t < o) red[t] += red[t + o]; __syncthreads(); }
  const float inv = 1.0f / red[0];
  for (int i = t; i < L; i += 256) x[i] *= inv;
}

__global__ void ctx_mean_kernel(const float* __restrict__ x, float* __restrict__ cm) {
  int d = blockIdx.x * 256 + threadIdx.x;
  if (d >= kD) return;
  float s = 0.0f;
  for (int b = 0; b < kB; ++b) s += x[((long)b * kS + (kS - 1)) * kD + d];
  cm[d] = kSqrtD * s / kB;
}

__global__ __launch_bounds__(256) void gate_kernel(
    const float* __restrict__ cm, const float* __restrict__ W,
    const float* __restrict__ bv, float* __restrict__ mean_w) {
  int n = blockIdx.x * 256 + threadIdx.x;
  if (n >= kNact) return;
  const float* wr = W + (long)n * kD;
  float s = 0.0f;
  for (int d = 0; d < kD; ++d) s += cm[d] * wr[d];
  mean_w[n] = s + bv[n];
}

__device__ inline unsigned rotl32(unsigned x, unsigned n) { return (x << n) | (x >> (32 - n)); }
__device__ inline void threefry2x32(unsigned k0, unsigned k1, unsigned c0, unsigned c1,
                                    unsigned& o0, unsigned& o1) {
  const unsigned R[8] = {13, 15, 26, 6, 17, 29, 16, 24};
  unsigned ks[3] = {k0, k1, k0 ^ k1 ^ 0x1BD11BDAu};
  unsigned x0 = c0 + ks[0], x1 = c1 + ks[1];
  #pragma unroll
  for (int i = 0; i < 5; ++i) {
    #pragma unroll
    for (int r = 0; r < 4; ++r) { x0 += x1; x1 = rotl32(x1, R[(i & 1) * 4 + r]); x1 ^= x0; }
    x0 += ks[(i + 1) % 3];
    x1 += ks[(i + 2) % 3] + (unsigned)(i + 1);
  }
  o0 = x0; o1 = x1;
}

__global__ __launch_bounds__(512) void gate_finalize_kernel(
    const float* __restrict__ mean_w, const int* __restrict__ uids,
    float* __restrict__ join_w, float* __restrict__ imp_out) {
  __shared__ float red[512];
  __shared__ float wsh[512];
  __shared__ float tot[1024];
  __shared__ float topv[kTopK];
  __shared__ int   topi;
  const int t = threadIdx.x;
  const float mv = mean_w[t];
  red[t] = mv; __syncthreads();
  for (int o = 256; o > 0; o >>= 1) { if (t < o) red[t] += red[t + o]; __syncthreads(); }
  const float mu = red[0] / kNact; __syncthreads();
  red[t] = (mv - mu) * (mv - mu); __syncthreads();
  for (int o = 256; o > 0; o >>= 1) { if (t < o) red[t] += red[t + o]; __syncthreads(); }
  const float stdv = sqrtf(red[0] / (kNact - 1)); __syncthreads();
  // gaussian noise: threefry2x32(key=(0,1), ctr=(0,t)) -> uniform -> erfinv
  unsigned r0, r1; threefry2x32(0u, 1u, 0u, (unsigned)t, r0, r1); (void)r1;
  float u = ((float)(r0 >> 8) + 0.5f) * (1.0f / 16777216.0f);
  u = fminf(fmaxf(u, 1e-6f), 1.0f - 1e-6f);
  const float wv = mv + stdv * 1.41421356237f * erfinvf(2.0f * u - 1.0f);
  // scatter into total_w[NTOT] then gather via uids
  tot[t] = 0.0f; tot[t + 512] = 0.0f; __syncthreads();
  tot[uids[t]] = wv; __syncthreads();
  const float aw = tot[uids[t]];
  wsh[t] = aw;
  // importance loss: IMP * (std(aw,ddof=1)/mean(aw))^2
  red[t] = aw; __syncthreads();
  for (int o = 256; o > 0; o >>= 1) { if (t < o) red[t] += red[t + o]; __syncthreads(); }
  const float am = red[0] / kNact; __syncthreads();
  red[t] = (aw - am) * (aw - am); __syncthreads();
  for (int o = 256; o > 0; o >>= 1) { if (t < o) red[t] += red[t + o]; __syncthreads(); }
  if (t == 0) imp_out[0] = kImp * (red[0] / (kNact - 1)) / (am * am);
  __syncthreads();
  // top-8 by iterated max + mask
  for (int kk = 0; kk < kTopK; ++kk) {
    red[t] = wsh[t]; __syncthreads();
    for (int o = 256; o > 0; o >>= 1) { if (t < o) red[t] = fmaxf(red[t], red[t + o]); __syncthreads(); }
    const float mx = red[0];
    if (t == 0) { topv[kk] = mx; topi = 512; }
    __syncthreads();
    if (wsh[t] == mx) atomicMin(&topi, t);
    __syncthreads();
    if (t == topi) wsh[t] = -3.0e38f;
    __syncthreads();
  }
  if (t == 0) {
    float e[kTopK], s = 0.0f;
    const float mx = topv[0];
    for (int i = 0; i < kTopK; ++i) { e[i] = __expf(topv[i] - mx); s += e[i]; }
    for (int i = 0; i < kTopK; ++i) join_w[i] = e[i] / s;
  }
}

__global__ void mix_kernel(const float* __restrict__ jw, const float* __restrict__ resp,
                           float* __restrict__ out, long n, long kstride) {
  long i = (long)blockIdx.x * 256 + threadIdx.x;
  if (i >= n) return;
  float s = 0.0f;
  #pragma unroll
  for (int k = 0; k < kTopK; ++k) s += jw[k] * resp[(long)k * kstride + i];
  out[i] = s;
}

__global__ __launch_bounds__(256) void ce_kernel(const float* __restrict__ logits,
                                                 const int* __restrict__ inputs,
                                                 float* __restrict__ ce_acc) {
  const int row = blockIdx.x;              // 0 .. B*(S-1)-1
  const int b = row / (kS - 1);
  const int s = row % (kS - 1);
  const float* x = logits + ((long)b * kS + s) * kV;
  const int tgt = inputs[b * kS + s + 1];
  __shared__ float rm[256], rs[256];
  const int t = threadIdx.x;
  float m = -3.0e38f, sum = 0.0f;
  for (int i = t; i < kV; i += 256) {
    float v = x[i];
    if (v > m) { sum = sum * __expf(m - v) + 1.0f; m = v; }
    else        sum += __expf(v - m);
  }
  rm[t] = m; rs[t] = sum; __syncthreads();
  for (int o = 128; o > 0; o >>= 1) {
    if (t < o) {
      float m2 = rm[t + o], s2 = rs[t + o];
      float M = fmaxf(rm[t], m2);
      rs[t] = rs[t] * __expf(rm[t] - M) + s2 * __expf(m2 - M);
      rm[t] = M;
    }
    __syncthreads();
  }
  if (t == 0) {
    float lse = rm[0] + __logf(rs[0]);
    atomicAdd(ce_acc, lse - x[tgt]);
  }
}

__global__ void zero_kernel(float* p) { p[0] = 0.0f; }

__global__ void loss_final_kernel(float* __restrict__ out, const float* __restrict__ ce_acc,
                                  const float* __restrict__ imp) {
  out[0] = ce_acc[0] / (float)(kB * (kS - 1)) + imp[0];
}

// ---------------------------------------------------------------------------
// Host-side orchestration
// ---------------------------------------------------------------------------
static inline void launch_gemm(hipStream_t st, const float* A, const float* Bm,
                               const float* bias, const float* R, float* C,
                               int M, int N, int K, int lda, int ldb, int ldc,
                               long sAb, long sAh, long sBb, long sBh, long sCb, long sCh,
                               int Hdiv, int Z, float alpha, int flags) {
  dim3 grid((N + TN - 1) / TN, (M + TM - 1) / TM, Z);
  gemm_wmma_kernel<<<grid, 256, 0, st>>>(A, Bm, bias, R, C, M, N, K, lda, ldb, ldc,
                                         sAb, sAh, sBb, sBh, sCb, sCh, Hdiv, alpha, flags);
}

static void run_layer(hipStream_t st, float* x, float* xmid, float* tbuf, float* qkv,
                      float* hbuf, float* scor, float* obuf,
                      const float* Wqkv, const float* bqkv, const float* Wo, const float* bo,
                      const float* W1, const float* b1, const float* W2, const float* b2,
                      const float* g1, const float* be1, const float* g2, const float* be2) {
  const float attn_scale = 0.08838834764831845f;  // 1/sqrt(128)
  // qkv = x @ Wqkv^T + bqkv
  launch_gemm(st, x, Wqkv, bqkv, nullptr, qkv, kM, 3 * kD, kD, kD, kD, 3 * kD,
              0, 0, 0, 0, 0, 0, 1, 1, 1.0f, 0);
  // scores[b,h] = scale * Q @ K^T   (z = b*H + h)
  launch_gemm(st, qkv, qkv + kD, nullptr, nullptr, scor, kS, kS, kD / kH,
              3 * kD, 3 * kD, kS,
              (long)kS * 3 * kD, kD / kH, (long)kS * 3 * kD, kD / kH,
              (long)kH * kS * kS, (long)kS * kS, kH, kB * kH, attn_scale, 0);
  // softmax rows
  softmax_kernel<<<dim3(kB * kH * kS), 256, 0, st>>>(scor, kS);
  // o[b,h] = P @ V    (V is K-major: rows = tokens)
  launch_gemm(st, scor, qkv + 2 * kD, nullptr, nullptr, obuf, kS, kD / kH, kS,
              kS, 3 * kD, kD,
              (long)kH * kS * kS, (long)kS * kS, (long)kS * 3 * kD, kD / kH,
              (long)kS * kD, kD / kH, kH, kB * kH, 1.0f, 4);
  // t = o @ Wo^T + bo + x (residual)
  launch_gemm(st, obuf, Wo, bo, x, tbuf, kM, kD, kD, kD, kD, kD,
              0, 0, 0, 0, 0, 0, 1, 1, 1.0f, 2);
  // xmid = LN(t)
  layernorm_kernel<<<dim3(kM), 256, 0, st>>>(tbuf, g1, be1, xmid);
  // h = relu(xmid @ W1^T + b1)
  launch_gemm(st, xmid, W1, b1, nullptr, hbuf, kM, kHid, kD, kD, kD, kHid,
              0, 0, 0, 0, 0, 0, 1, 1, 1.0f, 1);
  // t = h @ W2^T + b2 + xmid
  launch_gemm(st, hbuf, W2, b2, xmid, tbuf, kM, kD, kHid, kHid, kHid, kD,
              0, 0, 0, 0, 0, 0, 1, 1, 1.0f, 2);
  // x = LN(t)
  layernorm_kernel<<<dim3(kM), 256, 0, st>>>(tbuf, g2, be2, x);
}

extern "C" void kernel_launch(void* const* d_in, const int* in_sizes, int n_in,
                              void* d_out, int out_size, void* d_ws, size_t ws_size,
                              hipStream_t stream) {
  (void)in_sizes; (void)n_in; (void)out_size; (void)ws_size;
  const int*   inputs = (const int*)d_in[0];
  const int*   uids   = (const int*)d_in[1];
  const float* resp   = (const float*)d_in[2];
  const float* emb    = (const float*)d_in[3];
  const float* gate_W = (const float*)d_in[4];
  const float* gate_b = (const float*)d_in[5];
  const float* dec_W  = (const float*)d_in[6];
  const float* loc[12]; for (int i = 0; i < 12; ++i) loc[i] = (const float*)d_in[7 + i];
  const float* enc[12]; for (int i = 0; i < 12; ++i) enc[i] = (const float*)d_in[19 + i];

  float* out = (float*)d_out;
  float* w = (float*)d_ws;
  size_t o = 0;
  float* xbuf = w + o; o += (size_t)kM * kD;
  float* xmid = w + o; o += (size_t)kM * kD;
  float* tbuf = w + o; o += (size_t)kM * kD;
  float* qkv  = w + o; o += (size_t)kM * 3 * kD;
  float* hbuf = w + o; o += (size_t)kM * kHid;
  float* scor = w + o; o += (size_t)kB * kH * kS * kS;
  float* obuf = w + o; o += (size_t)kM * kD;
  float* cm   = w + o; o += kD;
  float* mw   = w + o; o += kNact;
  float* jw   = w + o; o += kTopK;
  float* imp  = w + o; o += 1;
  float* ceac = w + o; o += 1;

  // 1. x = emb[inputs]
  {
    long n = (long)kM * kD;
    embed_kernel<<<dim3((unsigned)((n + 255) / 256)), 256, 0, stream>>>(inputs, emb, xbuf);
  }
  // 2. local encoder layer
  run_layer(stream, xbuf, xmid, tbuf, qkv, hbuf, scor, obuf,
            loc[0], loc[1], loc[2], loc[3], loc[4], loc[5],
            loc[6], loc[7], loc[8], loc[9], loc[10], loc[11]);
  // 3-5. ctx mean (scaled by sqrt(D)); gate GEMV; noise/topk/softmax/imp
  ctx_mean_kernel<<<dim3((kD + 255) / 256), 256, 0, stream>>>(xbuf, cm);
  gate_kernel<<<dim3((kNact + 255) / 256), 256, 0, stream>>>(cm, gate_W, gate_b, mw);
  gate_finalize_kernel<<<1, 512, 0, stream>>>(mw, uids, jw, imp);
  // 6. out = sum_k join_w[k] * responses[k]
  {
    long n = (long)kM * kD;
    mix_kernel<<<dim3((unsigned)((n + 255) / 256)), 256, 0, stream>>>(jw, resp, xbuf, n, n);
  }
  // 7. two encoder layers
  for (int L = 0; L < 2; ++L) {
    run_layer(stream, xbuf, xmid, tbuf, qkv, hbuf, scor, obuf,
              enc[0] + (size_t)L * 3 * kD * kD, enc[1] + (size_t)L * 3 * kD,
              enc[2] + (size_t)L * kD * kD,     enc[3] + (size_t)L * kD,
              enc[4] + (size_t)L * kHid * kD,   enc[5] + (size_t)L * kHid,
              enc[6] + (size_t)L * kD * kHid,   enc[7] + (size_t)L * kD,
              enc[8] + (size_t)L * kD,          enc[9] + (size_t)L * kD,
              enc[10] + (size_t)L * kD,         enc[11] + (size_t)L * kD);
  }
  // 8. logits = x @ dec_W^T   (straight into d_out[1:])
  launch_gemm(stream, xbuf, dec_W, nullptr, nullptr, out + 1,
              kM, kV, kD, kD, kD, kV, 0, 0, 0, 0, 0, 0, 1, 1, 1.0f, 0);
  // 9. cross-entropy + importance -> d_out[0]
  zero_kernel<<<1, 1, 0, stream>>>(ceac);
  ce_kernel<<<dim3(kB * (kS - 1)), 256, 0, stream>>>(out + 1, inputs, ceac);
  loss_final_kernel<<<1, 1, 0, stream>>>(out, ceac, imp);
}
